// SiMBA_67078799229635
// MI455X (gfx1250) — matrix operand order repaired and checked by reading, not compiled, not run on hardware
//
#include <hip/hip_runtime.h>
#include <math.h>

// ---------------- types ----------------
typedef __bf16 bf16_t;
typedef bf16_t v16bf __attribute__((ext_vector_type(16)));
typedef float  v8f   __attribute__((ext_vector_type(8)));
typedef unsigned short u16;

// problem constants
#define BB   2
#define LN   784          // tokens
#define CC   384          // channels
#define DIN  768          // d_inner
#define DST  64           // d_state
#define DTR  24           // dt_rank
#define ROWS (BB*LN)      // 1568
#define NBK  4
#define BSZ  96
#define NPAD 800          // 784 padded to mult of 32

__device__ __forceinline__ u16 f2bf(float f) {
    unsigned int u = __builtin_bit_cast(unsigned int, f);
    u += 0x7fffu + ((u >> 16) & 1u);      // round-to-nearest-even
    return (u16)(u >> 16);
}
__device__ __forceinline__ float shrinkf(float v) {
    float a = fabsf(v) - 0.01f;
    a = a > 0.f ? a : 0.f;
    return v < 0.f ? -a : a;
}

// fragment loaders (ISA 7.12.2 layouts)
__device__ __forceinline__ v16bf load_afrag(const u16* __restrict__ A, int K,
                                            int m, int k0, int kh) {
    v16bf af;
    const u16* pa = A + (long)m * K + k0 + kh * 8;
#pragma unroll
    for (int j = 0; j < 8; ++j) {
        af[j]     = __builtin_bit_cast(bf16_t, pa[j]);
        af[j + 8] = __builtin_bit_cast(bf16_t, pa[j + 16]);
    }
    return af;
}
__device__ __forceinline__ v16bf load_bfrag(const u16* __restrict__ Bt, int K,
                                            int n, int k0, int kh) {
    v16bf bfv;
    const u16* pb = Bt + (long)n * K + k0 + kh * 8;
#pragma unroll
    for (int j = 0; j < 8; ++j) {
        bfv[j]     = __builtin_bit_cast(bf16_t, pb[j]);
        bfv[j + 8] = __builtin_bit_cast(bf16_t, pb[j + 16]);
    }
    return bfv;
}

// ---------------- generic bf16 WMMA GEMM (1 tile / wave, batched) ----------------
// C[m,n] = sum_k A[m,k]*Bt[n,k]  (+ beta*Cadd[m,n])
__global__ void k_gemm(const u16* __restrict__ A, const u16* __restrict__ Bt,
                       float* __restrict__ C, int M, int N, int K,
                       const float* __restrict__ Cadd, float beta,
                       long sA, long sB, long sC, long sAdd)
{
    int z = blockIdx.z;
    A  += z * sA;  Bt += z * sB;  C += z * sC;
    const float* add = Cadd ? (Cadd + z * sAdd) : nullptr;
    int nt = blockIdx.x, mt = blockIdx.y;
    int lane = threadIdx.x & 31;
    int loc  = lane & 15;
    int kh   = lane >> 4;
    int m = mt * 16 + loc;
    int n = nt * 16 + loc;

    v8f acc;
#pragma unroll
    for (int i = 0; i < 8; ++i) acc[i] = 0.f;

    for (int k0 = 0; k0 < K; k0 += 32) {
        v16bf af  = load_afrag(A, K, m, k0, kh);
        v16bf bfv = load_bfrag(Bt, K, n, k0, kh);
        acc = __builtin_amdgcn_wmma_f32_16x16x32_bf16(
                  false, af, false, bfv, (short)0, acc, false, false);
    }
#pragma unroll
    for (int v = 0; v < 8; ++v) {
        int mr = mt * 16 + kh * 8 + v;
        long o = (long)mr * N + n;
        float r = acc[v];
        if (add) r += beta * add[o];
        C[o] = r;
    }
}

// ---------------- 16x64 macro-tile WMMA GEMM (A-fragment reuse, N%64==0) --------
// One wave computes 4 adjacent 16x16 tiles along N: 1 A load + 4 B loads -> 4 WMMA.
__global__ void k_gemm4(const u16* __restrict__ A, const u16* __restrict__ Bt,
                        float* __restrict__ C, int M, int N, int K,
                        const float* __restrict__ Cadd, float beta)
{
    int nt0 = blockIdx.x * 4, mt = blockIdx.y;
    int lane = threadIdx.x & 31;
    int loc  = lane & 15;
    int kh   = lane >> 4;
    int m = mt * 16 + loc;

    v8f acc0, acc1, acc2, acc3;
#pragma unroll
    for (int i = 0; i < 8; ++i) { acc0[i]=0.f; acc1[i]=0.f; acc2[i]=0.f; acc3[i]=0.f; }

    int n0 = (nt0 + 0) * 16 + loc;
    int n1 = (nt0 + 1) * 16 + loc;
    int n2 = (nt0 + 2) * 16 + loc;
    int n3 = (nt0 + 3) * 16 + loc;

    for (int k0 = 0; k0 < K; k0 += 32) {
        v16bf af = load_afrag(A, K, m, k0, kh);
        v16bf b0 = load_bfrag(Bt, K, n0, k0, kh);
        v16bf b1 = load_bfrag(Bt, K, n1, k0, kh);
        v16bf b2 = load_bfrag(Bt, K, n2, k0, kh);
        v16bf b3 = load_bfrag(Bt, K, n3, k0, kh);
        acc0 = __builtin_amdgcn_wmma_f32_16x16x32_bf16(false, af, false, b0, (short)0, acc0, false, false);
        acc1 = __builtin_amdgcn_wmma_f32_16x16x32_bf16(false, af, false, b1, (short)0, acc1, false, false);
        acc2 = __builtin_amdgcn_wmma_f32_16x16x32_bf16(false, af, false, b2, (short)0, acc2, false, false);
        acc3 = __builtin_amdgcn_wmma_f32_16x16x32_bf16(false, af, false, b3, (short)0, acc3, false, false);
    }
#pragma unroll
    for (int v = 0; v < 8; ++v) {
        int mr = mt * 16 + kh * 8 + v;
        long rb = (long)mr * N;
        float r0 = acc0[v], r1 = acc1[v], r2 = acc2[v], r3 = acc3[v];
        if (Cadd) {
            r0 += beta * Cadd[rb + n0];
            r1 += beta * Cadd[rb + n1];
            r2 += beta * Cadd[rb + n2];
            r3 += beta * Cadd[rb + n3];
        }
        C[rb + n0] = r0;
        C[rb + n1] = r1;
        C[rb + n2] = r2;
        C[rb + n3] = r3;
    }
}

// ---------------- elementwise / prep kernels ----------------
__global__ void k_pad_convert(const float* __restrict__ src, long srcStride,
                              u16* __restrict__ dst, int R, int C, int Rp, int Cp)
{
    int i = blockIdx.x * blockDim.x + threadIdx.x;
    if (i >= Rp * Cp) return;
    int c = i % Cp, r = i / Cp;
    dst[i] = (r < R && c < C) ? f2bf(src[(long)r * srcStride + c]) : (u16)0;
}

__global__ void k_cwt(const float* __restrict__ cw, u16* __restrict__ dst)
{
    int i = blockIdx.x * blockDim.x + threadIdx.x;
    if (i >= 2 * 4 * 96 * 96) return;
    int d = i % 96, t = i / 96;
    int kk = t % 96, g = t / 96;
    dst[(long)g * 9216 + (long)kk * 96 + d] = f2bf(cw[(long)g * 9216 + (long)d * 96 + kk]);
}

// DFT twiddles (shared fwd/inv; cos and -sin are symmetric in (k,n)): [784,800] bf16
__global__ void k_gen_dft(u16* __restrict__ Wc, u16* __restrict__ Ws)
{
    int i = blockIdx.x * blockDim.x + threadIdx.x;
    if (i >= LN * NPAD) return;
    int n = i % NPAD, k = i / NPAD;
    if (n >= LN) { Wc[i] = 0; Ws[i] = 0; return; }
    int m = (int)(((long)k * n) % LN);
    float th = 6.28318530717958647692f * (float)m / (float)LN;
    Wc[i] = f2bf(cosf(th));
    Ws[i] = f2bf(-sinf(th));
}

__global__ void k_layernorm(const float* __restrict__ x, const float* __restrict__ w,
                            const float* __restrict__ b, float* __restrict__ outf,
                            u16* __restrict__ outbf)
{
    int row = blockIdx.x, tid = threadIdx.x;
    const float* xr = x + (long)row * CC;
    __shared__ float s1[128], s2[128];
    float s = 0.f, q = 0.f;
    for (int c = tid; c < CC; c += 128) { float v = xr[c]; s += v; q += v * v; }
    s1[tid] = s; s2[tid] = q; __syncthreads();
    for (int off = 64; off > 0; off >>= 1) {
        if (tid < off) { s1[tid] += s1[tid + off]; s2[tid] += s2[tid + off]; }
        __syncthreads();
    }
    float mu  = s1[0] / CC;
    float var = s2[0] / CC - mu * mu;
    float inv = rsqrtf(var + 1e-5f);
    for (int c = tid; c < CC; c += 128) {
        float v = (xr[c] - mu) * inv * w[c] + b[c];
        if (outf)  outf[(long)row * CC + c]  = v;
        if (outbf) outbf[(long)row * CC + c] = f2bf(v);
    }
}

__global__ void k_conv_silu(const float* __restrict__ xz, const float* __restrict__ cw,
                            const float* __restrict__ cb, float* __restrict__ xc,
                            u16* __restrict__ xcbf)
{
    int i = blockIdx.x * blockDim.x + threadIdx.x;
    if (i >= ROWS * DIN) return;
    int d = i % DIN, r = i / DIN;
    int b = r / LN, l = r % LN;
    float acc = cb[d];
#pragma unroll
    for (int j = 0; j < 4; ++j) {
        int ll = l - 3 + j;
        if (ll >= 0) acc += cw[d * 4 + j] * xz[((long)b * LN + ll) * (2 * DIN) + d];
    }
    float s = acc / (1.f + __expf(-acc));
    xc[i]   = s;
    xcbf[i] = f2bf(s);
}

__global__ void k_dt_softplus(float* __restrict__ dt, const float* __restrict__ bias)
{
    int i = blockIdx.x * blockDim.x + threadIdx.x;
    if (i >= ROWS * DIN) return;
    float v = dt[i] + bias[i % DIN];
    dt[i] = (v > 20.f) ? v : log1pf(__expf(v));
}

// selective scan: one wave32 per (b, d_inner); 2 states/lane, shfl-xor reduce.
__global__ void k_scan(const float* __restrict__ dt, const float* __restrict__ dbl,
                       const float* __restrict__ xc, const float* __restrict__ xz,
                       const float* __restrict__ A_log, const float* __restrict__ Dp,
                       u16* __restrict__ ybf)
{
    int wid  = blockIdx.x * (blockDim.x >> 5) + (threadIdx.x >> 5);
    if (wid >= BB * DIN) return;
    int lane = threadIdx.x & 31;
    int b = wid / DIN, d = wid % DIN;
    float a0 = -__expf(A_log[d * DST + lane]);
    float a1 = -__expf(A_log[d * DST + 32 + lane]);
    float h0 = 0.f, h1 = 0.f;
    float Dd = Dp[d];
    for (int t = 0; t < LN; ++t) {
        long row = (long)b * LN + t;
        float dtv = dt[row * DIN + d];
        float u   = xc[row * DIN + d];
        float B0  = dbl[row * 160 + DTR + lane];
        float B1  = dbl[row * 160 + DTR + 32 + lane];
        float C0  = dbl[row * 160 + DTR + DST + lane];
        float C1  = dbl[row * 160 + DTR + DST + 32 + lane];
        h0 = h0 * __expf(dtv * a0) + dtv * B0 * u;
        h1 = h1 * __expf(dtv * a1) + dtv * B1 * u;
        float p = h0 * C0 + h1 * C1;
        for (int off = 16; off > 0; off >>= 1) p += __shfl_xor(p, off, 32);
        if (lane == 0) {
            float y = p + u * Dd;
            float z = xz[row * (2 * DIN) + DIN + d];
            y *= z / (1.f + __expf(-z));
            ybf[row * DIN + d] = f2bf(y);
        }
    }
}

// transpose+pad LN2 output to DFT B-operand: xt[(b*384+c)*800 + n]
__global__ void k_xt(const float* __restrict__ xln2, u16* __restrict__ xt)
{
    int i = blockIdx.x * blockDim.x + threadIdx.x;
    if (i >= BB * CC * NPAD) return;
    int n = i % NPAD, col = i / NPAD;
    int b = col / CC, c = col % CC;
    xt[i] = (n < LN) ? f2bf(xln2[((long)b * LN + n) * CC + c]) : (u16)0;
}

// forward radix-4 DFT over nb axis + ortho scale; write per-block bf16 A operands
__global__ void k_fft4_fwd(const float* __restrict__ Fr, const float* __restrict__ Fi,
                           u16* __restrict__ XFr, u16* __restrict__ XFi)
{
    int i = blockIdx.x * blockDim.x + threadIdx.x;
    if (i >= LN * BB * BSZ) return;
    int j = i % BSZ, t = i / BSZ;
    int b = t % BB, k = t / BB;
    float r[4], m[4];
#pragma unroll
    for (int q = 0; q < 4; ++q) {
        long idx = (long)k * (BB * CC) + (long)b * CC + q * BSZ + j;
        r[q] = Fr[idx]; m[q] = Fi[idx];
    }
    const float sc = 1.f / 56.f;          // 1/sqrt(784*4)
    float yr[4], yi[4];
    yr[0] = r[0]+r[1]+r[2]+r[3];  yi[0] = m[0]+m[1]+m[2]+m[3];
    yr[1] = r[0]+m[1]-r[2]-m[3];  yi[1] = m[0]-r[1]-m[2]+r[3];
    yr[2] = r[0]-r[1]+r[2]-r[3];  yi[2] = m[0]-m[1]+m[2]-m[3];
    yr[3] = r[0]-m[1]-r[2]+m[3];  yi[3] = m[0]+r[1]-m[2]-r[3];
    long rrow = (long)b * LN + k;
#pragma unroll
    for (int p = 0; p < 4; ++p) {
        long o = (long)p * (ROWS * BSZ) + rrow * BSZ + j;
        XFr[o] = f2bf(sc * yr[p]);
        XFi[o] = f2bf(sc * yi[p]);
    }
}

__global__ void k_biasrelu(const float* __restrict__ r1, const float* __restrict__ i1,
                           const float* __restrict__ cb1,
                           u16* __restrict__ r1bf, u16* __restrict__ i1bf)
{
    int i = blockIdx.x * blockDim.x + threadIdx.x;
    if (i >= NBK * ROWS * BSZ) return;
    int j = i % BSZ, blk = i / (ROWS * BSZ);
    float rv = r1[i] + cb1[blk * BSZ + j];
    float iv = i1[i] + cb1[NBK * BSZ + blk * BSZ + j];
    r1bf[i] = f2bf(rv > 0.f ? rv : 0.f);
    i1bf[i] = f2bf(iv > 0.f ? iv : 0.f);
}

// bias + soft-shrink + inverse radix-4 over nb + transpose/pad to inverse-DFT operand
__global__ void k_shrink_ifft4(const float* __restrict__ r2, const float* __restrict__ i2,
                               const float* __restrict__ cb2,
                               u16* __restrict__ Yr, u16* __restrict__ Yi)
{
    int i = blockIdx.x * blockDim.x + threadIdx.x;
    if (i >= BB * BSZ * NPAD) return;
    int k = i % NPAD, t = i / NPAD;
    int j = t % BSZ, b = t / BSZ;
    float yr[4] = {0,0,0,0}, yi[4] = {0,0,0,0};
    if (k < LN) {
        float r[4], m[4];
        long rrow = (long)b * LN + k;
#pragma unroll
        for (int q = 0; q < 4; ++q) {
            long o = (long)q * (ROWS * BSZ) + rrow * BSZ + j;
            r[q] = shrinkf(r2[o] + cb2[q * BSZ + j]);
            m[q] = shrinkf(i2[o] + cb2[NBK * BSZ + q * BSZ + j]);
        }
        // inverse DFT-4: e^{+2*pi*i*p*q/4}
        yr[0] = r[0]+r[1]+r[2]+r[3];  yi[0] = m[0]+m[1]+m[2]+m[3];
        yr[1] = r[0]-m[1]-r[2]+m[3];  yi[1] = m[0]+r[1]-m[2]-r[3];
        yr[2] = r[0]-r[1]+r[2]-r[3];  yi[2] = m[0]-m[1]+m[2]-m[3];
        yr[3] = r[0]+m[1]-r[2]-m[3];  yi[3] = m[0]-r[1]-m[2]+r[3];
    }
    const float sc = 1.f / 56.f;          // inverse ortho scale
#pragma unroll
    for (int p = 0; p < 4; ++p) {
        long o = ((long)b * CC + p * BSZ + j) * NPAD + k;
        Yr[o] = f2bf(sc * yr[p]);
        Yi[o] = f2bf(sc * yi[p]);
    }
}

__global__ void k_final(const float* __restrict__ x1, const float* __restrict__ T,
                        float* __restrict__ out)
{
    int i = blockIdx.x * blockDim.x + threadIdx.x;
    if (i >= ROWS * CC) return;
    int c = i % CC, row = i / CC;
    int b = row / LN, n = row % LN;
    out[i] = x1[i] + T[(long)n * (BB * CC) + (long)b * CC + c];
}

// ---------------- host orchestration ----------------
static inline int cd(int a, int b) { return (a + b - 1) / b; }

extern "C" void kernel_launch(void* const* d_in, const int* in_sizes, int n_in,
                              void* d_out, int out_size, void* d_ws, size_t ws_size,
                              hipStream_t stream)
{
    (void)in_sizes; (void)n_in; (void)out_size; (void)ws_size;
    const float* x     = (const float*)d_in[0];
    const float* ln1w  = (const float*)d_in[3];
    const float* ln1b  = (const float*)d_in[4];
    const float* w_in  = (const float*)d_in[5];
    const float* convw = (const float*)d_in[6];
    const float* convb = (const float*)d_in[7];
    const float* w_xp  = (const float*)d_in[8];
    const float* w_dt  = (const float*)d_in[9];
    const float* dtb   = (const float*)d_in[10];
    const float* A_log = (const float*)d_in[11];
    const float* Dp    = (const float*)d_in[12];
    const float* w_out = (const float*)d_in[13];
    const float* ln2w  = (const float*)d_in[14];
    const float* ln2b  = (const float*)d_in[15];
    const float* cw1   = (const float*)d_in[16];
    const float* cb1   = (const float*)d_in[17];
    const float* cw2   = (const float*)d_in[18];
    const float* cb2   = (const float*)d_in[19];
    float* out = (float*)d_out;

    char* base = (char*)d_ws;
    size_t off = 0;
    auto alloc = [&](size_t bytes) -> void* {
        void* p = base + off;
        off = (off + bytes + 255) & ~(size_t)255;
        return p;
    };
    u16*   w_in_bf  = (u16*)alloc((size_t)1536 * 384 * 2);
    u16*   w_xp_bf  = (u16*)alloc((size_t)160 * 768 * 2);
    u16*   w_dt_bf  = (u16*)alloc((size_t)768 * 32 * 2);
    u16*   w_out_bf = (u16*)alloc((size_t)384 * 768 * 2);
    u16*   cw1t     = (u16*)alloc((size_t)2 * 4 * 96 * 96 * 2);
    u16*   cw2t     = (u16*)alloc((size_t)2 * 4 * 96 * 96 * 2);
    u16*   Wc       = (u16*)alloc((size_t)LN * NPAD * 2);
    u16*   Ws       = (u16*)alloc((size_t)LN * NPAD * 2);
    u16*   xln_bf   = (u16*)alloc((size_t)ROWS * CC * 2);
    float* xz       = (float*)alloc((size_t)ROWS * 2 * DIN * 4);
    float* xc       = (float*)alloc((size_t)ROWS * DIN * 4);
    u16*   xc_bf    = (u16*)alloc((size_t)ROWS * DIN * 2);
    float* dbl      = (float*)alloc((size_t)ROWS * 160 * 4);
    u16*   dt_bf    = (u16*)alloc((size_t)ROWS * 32 * 2);
    float* dtf      = (float*)alloc((size_t)ROWS * DIN * 4);
    u16*   ybf      = (u16*)alloc((size_t)ROWS * DIN * 2);
    float* x1       = (float*)alloc((size_t)ROWS * CC * 4);
    float* xln2     = (float*)alloc((size_t)ROWS * CC * 4);
    u16*   xt_bf    = (u16*)alloc((size_t)BB * CC * NPAD * 2);
    float* Fr       = (float*)alloc((size_t)LN * BB * CC * 4);  // reused as tmpA
    float* Fi       = (float*)alloc((size_t)LN * BB * CC * 4);  // reused as tmpB
    u16*   XFr_bf   = (u16*)alloc((size_t)NBK * ROWS * BSZ * 2);
    u16*   XFi_bf   = (u16*)alloc((size_t)NBK * ROWS * BSZ * 2);
    float* r1pre    = (float*)alloc((size_t)NBK * ROWS * BSZ * 4); // reused as r2pre
    float* i1pre    = (float*)alloc((size_t)NBK * ROWS * BSZ * 4); // reused as i2pre
    u16*   r1bf     = (u16*)alloc((size_t)NBK * ROWS * BSZ * 2);
    u16*   i1bf     = (u16*)alloc((size_t)NBK * ROWS * BSZ * 2);
    u16*   Yr_bf    = (u16*)alloc((size_t)BB * CC * NPAD * 2);
    u16*   Yi_bf    = (u16*)alloc((size_t)BB * CC * NPAD * 2);
    float* T1       = (float*)alloc((size_t)LN * BB * CC * 4);
    float* T        = (float*)alloc((size_t)LN * BB * CC * 4);
    float* tmpA = Fr;      // Fr/Fi dead after k_fft4_fwd
    float* tmpB = Fi;
    float* r2pre = r1pre;  // r1pre/i1pre dead after k_biasrelu
    float* i2pre = i1pre;

    const int EB = 256;
    // ---- weight prep ----
    k_pad_convert<<<cd(1536*384,EB),EB,0,stream>>>(w_in, 384, w_in_bf, 1536, 384, 1536, 384);
    k_pad_convert<<<cd(160*768,EB),EB,0,stream>>>(w_xp, 768, w_xp_bf, 152, 768, 160, 768);
    k_pad_convert<<<cd(768*32,EB),EB,0,stream>>>(w_dt, 24, w_dt_bf, 768, 24, 768, 32);
    k_pad_convert<<<cd(384*768,EB),EB,0,stream>>>(w_out, 768, w_out_bf, 384, 768, 384, 768);
    k_cwt<<<cd(2*4*96*96,EB),EB,0,stream>>>(cw1, cw1t);
    k_cwt<<<cd(2*4*96*96,EB),EB,0,stream>>>(cw2, cw2t);
    k_gen_dft<<<cd(LN*NPAD,EB),EB,0,stream>>>(Wc, Ws);

    // ---- Mamba branch ----
    k_layernorm<<<ROWS,128,0,stream>>>(x, ln1w, ln1b, nullptr, xln_bf);
    k_gemm4<<<dim3(1536/64, ROWS/16),32,0,stream>>>(xln_bf, w_in_bf, xz,
            ROWS, 1536, 384, nullptr, 0.f);
    k_conv_silu<<<cd(ROWS*DIN,EB),EB,0,stream>>>(xz, convw, convb, xc, xc_bf);
    k_gemm<<<dim3(160/16, ROWS/16, 1),32,0,stream>>>(xc_bf, w_xp_bf, dbl,
            ROWS, 160, 768, nullptr, 0.f, 0,0,0,0);
    k_pad_convert<<<cd(ROWS*32,EB),EB,0,stream>>>(dbl, 160, dt_bf, ROWS, DTR, ROWS, 32);
    k_gemm4<<<dim3(768/64, ROWS/16),32,0,stream>>>(dt_bf, w_dt_bf, dtf,
            ROWS, 768, 32, nullptr, 0.f);
    k_dt_softplus<<<cd(ROWS*DIN,EB),EB,0,stream>>>(dtf, dtb);
    k_scan<<<(BB*DIN)/4,128,0,stream>>>(dtf, dbl, xc, xz, A_log, Dp, ybf);
    k_gemm4<<<dim3(384/64, ROWS/16),32,0,stream>>>(ybf, w_out_bf, x1,
            ROWS, 384, 768, x, 1.f);   // + residual

    // ---- EinFFT branch ----
    k_layernorm<<<ROWS,128,0,stream>>>(x1, ln2w, ln2b, xln2, nullptr);
    k_xt<<<cd(BB*CC*NPAD,EB),EB,0,stream>>>(xln2, xt_bf);
    // forward DFT over token axis (as GEMM): Fr = Wc @ xt, Fi = Ws @ xt
    k_gemm4<<<dim3((BB*CC)/64, LN/16),32,0,stream>>>(Wc, xt_bf, Fr,
            LN, BB*CC, NPAD, nullptr, 0.f);
    k_gemm4<<<dim3((BB*CC)/64, LN/16),32,0,stream>>>(Ws, xt_bf, Fi,
            LN, BB*CC, NPAD, nullptr, 0.f);
    k_fft4_fwd<<<cd(LN*BB*BSZ,EB),EB,0,stream>>>(Fr, Fi, XFr_bf, XFi_bf);

    const long sAB = (long)ROWS * BSZ, sW = 9216;
    dim3 gblk(BSZ/16, ROWS/16, NBK);
    // layer 1:  r1 = Re@W0 - Im@W1 ; i1 = Re@W1 + Im@W0   (batched over 4 blocks)
    k_gemm<<<gblk,32,0,stream>>>(XFi_bf, cw1t + 4*sW, tmpA,  ROWS, BSZ, BSZ, nullptr, 0.f, sAB,sW,sAB,0);
    k_gemm<<<gblk,32,0,stream>>>(XFr_bf, cw1t,        r1pre, ROWS, BSZ, BSZ, tmpA, -1.f, sAB,sW,sAB,sAB);
    k_gemm<<<gblk,32,0,stream>>>(XFi_bf, cw1t,        tmpB,  ROWS, BSZ, BSZ, nullptr, 0.f, sAB,sW,sAB,0);
    k_gemm<<<gblk,32,0,stream>>>(XFr_bf, cw1t + 4*sW, i1pre, ROWS, BSZ, BSZ, tmpB,  1.f, sAB,sW,sAB,sAB);
    k_biasrelu<<<cd(NBK*ROWS*BSZ,EB),EB,0,stream>>>(r1pre, i1pre, cb1, r1bf, i1bf);
    // layer 2
    k_gemm<<<gblk,32,0,stream>>>(i1bf, cw2t + 4*sW, tmpA,  ROWS, BSZ, BSZ, nullptr, 0.f, sAB,sW,sAB,0);
    k_gemm<<<gblk,32,0,stream>>>(r1bf, cw2t,        r2pre, ROWS, BSZ, BSZ, tmpA, -1.f, sAB,sW,sAB,sAB);
    k_gemm<<<gblk,32,0,stream>>>(i1bf, cw2t,        tmpB,  ROWS, BSZ, BSZ, nullptr, 0.f, sAB,sW,sAB,0);
    k_gemm<<<gblk,32,0,stream>>>(r1bf, cw2t + 4*sW, i2pre, ROWS, BSZ, BSZ, tmpB,  1.f, sAB,sW,sAB,sAB);
    k_shrink_ifft4<<<cd(BB*BSZ*NPAD,EB),EB,0,stream>>>(r2pre, i2pre, cb2, Yr_bf, Yi_bf);
    // inverse DFT over token axis, real part: T = Wc@Yr + Ws@Yi  (Ws holds -sin)
    k_gemm4<<<dim3((BB*CC)/64, LN/16),32,0,stream>>>(Wc, Yr_bf, T1,
            LN, BB*CC, NPAD, nullptr, 0.f);
    k_gemm4<<<dim3((BB*CC)/64, LN/16),32,0,stream>>>(Ws, Yi_bf, T,
            LN, BB*CC, NPAD, T1, 1.f);
    k_final<<<cd(ROWS*CC,EB),EB,0,stream>>>(x1, T, out);
}
